// Split_BN_ReLU_DConv_27049704030775
// MI455X (gfx1250) — matrix operand order repaired and checked by reading, not compiled, not run on hardware
//
#include <hip/hip_runtime.h>
#include <hip/hip_bf16.h>

typedef _Float16 f16;
typedef __attribute__((ext_vector_type(16))) _Float16 v16h;
typedef __attribute__((ext_vector_type(8)))  _Float16 v8h;
typedef __attribute__((ext_vector_type(8)))  float    v8f;

#define BB      8
#define HH      64
#define WW      64
#define CIN     256
#define COUT    64
#define NSPLIT  4
#define MOFF    72       // 4 splits * 18 offset channels
#define MOFFP   80       // padded to 5 WMMA M-tiles
#define KOFF    2304     // 9 taps * 256 channels (k = tap*256 + c)
#define KDC     576      // 9 taps * 64 channels  (k = tap*64 + c)

// ---------------- workspace layout (bytes) ----------------
// h (NHWC f16)  : 8*64*64*256*2       = 16,777,216
// offsets (f32) : 8*72*64*64*4        =  9,437,184
// A1p (f16)     : 5*72*32*16*2        =    368,640   (per-lane fragment order)
// A2p (f16)     : 16*18*32*16*2       =    294,912   (per-lane fragment order)
// obias (f32)   : 80*4, bsum: 64*4
#define WS_H    0
#define WS_OFF  16777216u
#define WS_A1   26214400u
#define WS_A2   26583040u
#define WS_OB   26877952u
#define WS_BS   26878272u

static __device__ __forceinline__ v8f wmma_f16(v16h a, v16h b, v8f c) {
    return __builtin_amdgcn_wmma_f32_16x16x32_f16(false, a, false, b,
                                                  (short)0, c, false, false);
}

// ---------------------------------------------------------------------------
// Kernel 1: BN + ReLU, NCHW fp32 -> NHWC f16, v16h (b128-pair) stores
// ---------------------------------------------------------------------------
__global__ void bn_relu_nhwc(const float* __restrict__ x,
                             const float* __restrict__ gam, const float* __restrict__ bet,
                             const float* __restrict__ mu,  const float* __restrict__ var,
                             f16* __restrict__ h)
{
    __shared__ float sS[CIN], sT[CIN];
    int tid = threadIdx.y * 64 + threadIdx.x;
    if (tid < CIN) {
        float inv = rsqrtf(var[tid] + 1e-5f);
        float s   = gam[tid] * inv;
        sS[tid]   = s;
        sT[tid]   = bet[tid] - mu[tid] * s;
    }
    __syncthreads();
    int by = blockIdx.x;           // b*64 + y
    int xx = threadIdx.x;          // 0..63
    int cq = threadIdx.y;          // 0..3 -> 64-channel group
    const float* xp = x + ((size_t)(by >> 6) * CIN * HH + (by & 63)) * WW + xx;
    f16* hp = h + (((size_t)by) * WW + xx) * CIN + cq * 64;
    for (int q = 0; q < 4; ++q) {
        v16h vv;
        #pragma unroll
        for (int j = 0; j < 16; ++j) {
            int c = cq * 64 + q * 16 + j;
            float v = xp[(size_t)c * (HH * WW)];
            vv[j] = (f16)fmaxf(v * sS[c] + sT[c], 0.0f);
        }
        *(v16h*)(hp + q * 16) = vv;
    }
}

// ---------------------------------------------------------------------------
// Kernel 2: repack weights to f16 in per-lane WMMA fragment order.
// A fragment (16-bit A 16x32): lane L holds row L&15,
//   halves j -> K = kk*32 + (L>>4)*16 + j
// Packed so a lane's whole fragment is one 32B load, wave = 1KB contiguous.
// ---------------------------------------------------------------------------
__global__ void pack_weights(const float* __restrict__ offw, const float* __restrict__ offb,
                             const float* __restrict__ dcw,  const float* __restrict__ dcb,
                             f16* __restrict__ A1p, f16* __restrict__ A2p,
                             float* __restrict__ obias, float* __restrict__ bsum)
{
    int idx = blockIdx.x * 256 + threadIdx.x;
    const int N1 = 5 * 72 * 32 * 16;    // 184320
    const int N2 = 16 * 18 * 32 * 16;   // 147456
    if (idx < N1) {
        int j  = idx & 15;
        int L  = (idx >> 4) & 31;
        int r3 = idx >> 9;
        int kk = r3 % 72, mt = r3 / 72;
        int sel = L >> 4, row = L & 15;
        int m = mt * 16 + row;
        int kloc = sel * 16 + j;
        int kval = kk * 32 + kloc;
        int tap = kval >> 8, c = kval & 255;
        float v = 0.0f;
        if (m < MOFF) v = offw[((size_t)(m * 256 + c)) * 9 + tap];   // (4,18,256,3,3)
        A1p[idx] = (f16)v;
    } else if (idx < N1 + N2) {
        int id = idx - N1;
        int j  = id & 15;
        int L  = (id >> 4) & 31;
        int r3 = id >> 9;
        int kk = r3 % 18, q = r3 / 18;       // q = i*4 + wv
        int i = q >> 2, wvv = q & 3;
        int sel = L >> 4;
        int m = wvv * 16 + (L & 15);
        int kloc = sel * 16 + j;
        int kval = kk * 32 + kloc;
        int tap = kval >> 6, c = kval & 63;
        int mi = i * 64 + m;
        A2p[id] = (f16)dcw[((size_t)(mi * 64 + c)) * 9 + tap];       // (4,64,64,3,3)
    } else if (idx < N1 + N2 + MOFFP) {
        int m = idx - N1 - N2;
        obias[m] = (m < MOFF) ? offb[m] : 0.0f;
    } else if (idx < N1 + N2 + MOFFP + COUT) {
        int m = idx - N1 - N2 - MOFFP;
        float s = 0.0f;
        for (int i = 0; i < NSPLIT; ++i) s += dcb[i * COUT + m];
        bsum[m] = s;
    }
}

// ---------------------------------------------------------------------------
// Kernel 3: offset conv implicit GEMM (M=80, N=32768, K=2304), WMMA f16.
// One 16x16 D tile per wave. Per tap: bounds/base computed once, then an
// unrolled 8-chunk stream of {b128x2 B load, b128x2 A load, v_wmma}.
// ---------------------------------------------------------------------------
__global__ void offconv_wmma(const f16* __restrict__ h, const f16* __restrict__ A1p,
                             const float* __restrict__ obias, float* __restrict__ offs)
{
    int wv  = threadIdx.x >> 5;
    int L   = threadIdx.x & 31;
    int sel = L >> 4;
    int lc  = L & 15;
    int nt  = blockIdx.x * 4 + wv;     // N tile = 16 pixels of one row
    int b   = nt >> 8;
    int rem = nt & 255;
    int y   = rem >> 2;
    int x0  = (rem & 3) << 4;
    const f16* ap = A1p + ((size_t)blockIdx.y * 72 * 32 + L) * 16;
    const f16* hb = h + (size_t)b * HH * WW * CIN + sel * 16;
    v8f acc = {};
    for (int tap = 0; tap < 9; ++tap) {
        int yy = y + tap / 3 - 1;
        int xx = x0 + lc + tap % 3 - 1;
        bool valid = ((unsigned)yy < (unsigned)HH) && ((unsigned)xx < (unsigned)WW);
        const f16* bp = hb + ((size_t)yy * WW + xx) * CIN;
        if (tap < 8) {  // prefetch next tap's im2col row (global_prefetch_b8)
            int yn = y + (tap + 1) / 3 - 1;
            int xn = x0 + lc + (tap + 1) % 3 - 1;
            __builtin_prefetch(hb + ((size_t)yn * WW + xn) * CIN, 0, 3);
        }
        #pragma unroll
        for (int cc = 0; cc < 8; ++cc) {
            v16h bfr = {};
            if (valid) bfr = *(const v16h*)(bp + cc * 32);
            v16h a = *(const v16h*)(ap + cc * 512);
            acc = wmma_f16(a, bfr, acc);
        }
        ap += 8 * 512;
    }
    // D layout: VGPR r, lane L -> M = r + sel*8, N = lc
    int xo = x0 + lc;
    #pragma unroll
    for (int r = 0; r < 8; ++r) {
        int mm = blockIdx.y * 16 + r + sel * 8;
        if (mm < MOFF)
            offs[(((size_t)b * MOFF + mm) * HH + y) * WW + xo] = acc[r] + obias[mm];
    }
}

// ---------------------------------------------------------------------------
// Kernel 4: deformable conv. 4 waves/block = the 4 COUT M-tiles of one N tile.
// Per (split, tap): cooperative bilinear build of a 16px x 64ch B panel in
// LDS (one coord computation per tap, v8h corner gathers), then two WMMAs
// consume it -> 2 wmma per barrier pair. All 4 splits accumulate in-register.
// ---------------------------------------------------------------------------
__global__ void dconv_wmma(const f16* __restrict__ h, const f16* __restrict__ A2p,
                           const float* __restrict__ offs, const float* __restrict__ bsum,
                           float* __restrict__ out)
{
    __shared__ __align__(32) f16 Bp[16 * 64];   // [pixel][channel-of-tap], 2KB
    int tid = threadIdx.x;
    int wv  = tid >> 5, L = tid & 31, sel = L >> 4, lc = L & 15;
    int nt  = blockIdx.x;
    int b   = nt >> 8;
    int rem = nt & 255;
    int y   = rem >> 2;
    int x0  = (rem & 3) << 4;
    int p   = tid & 15;       // builder pixel (0..15)
    int g   = tid >> 4;       // builder channel group (0..7), 8 channels each
    v8f acc = {};
    for (int i = 0; i < NSPLIT; ++i) {
        const f16* ap = A2p + ((size_t)(i * 4 + wv) * 18 * 32 + L) * 16;
        for (int tap = 0; tap < 9; ++tap) {
            // ---- cooperative B panel build (bilinear sampling, whole tap) ----
            {
                int xs = x0 + p;
                int my = i * 18 + tap * 2;
                size_t ob = (((size_t)b * MOFF + my) * HH + y) * WW + xs;
                float py = (float)(y + tap / 3 - 1) + offs[ob];
                float px = (float)(xs + tap % 3 - 1) + offs[ob + (size_t)HH * WW];
                float fy = floorf(py), fx = floorf(px);
                float wy1 = py - fy, wx1 = px - fx;
                float wy0 = 1.0f - wy1, wx0 = 1.0f - wx1;
                int y0i = (int)fy, x0i = (int)fx;
                int y1i = y0i + 1, x1i = x0i + 1;
                float v0y = ((unsigned)y0i < (unsigned)HH) ? 1.0f : 0.0f;
                float v1y = ((unsigned)y1i < (unsigned)HH) ? 1.0f : 0.0f;
                float v0x = ((unsigned)x0i < (unsigned)WW) ? 1.0f : 0.0f;
                float v1x = ((unsigned)x1i < (unsigned)WW) ? 1.0f : 0.0f;
                int yc0 = y0i < 0 ? 0 : (y0i > HH - 1 ? HH - 1 : y0i);
                int yc1 = y1i < 0 ? 0 : (y1i > HH - 1 ? HH - 1 : y1i);
                int xc0 = x0i < 0 ? 0 : (x0i > WW - 1 ? WW - 1 : x0i);
                int xc1 = x1i < 0 ? 0 : (x1i > WW - 1 ? WW - 1 : x1i);
                const f16* base = h + (size_t)b * HH * WW * CIN + (i * COUT + g * 8);
                v8h a00 = *(const v8h*)(base + ((size_t)yc0 * WW + xc0) * CIN);
                v8h a01 = *(const v8h*)(base + ((size_t)yc0 * WW + xc1) * CIN);
                v8h a10 = *(const v8h*)(base + ((size_t)yc1 * WW + xc0) * CIN);
                v8h a11 = *(const v8h*)(base + ((size_t)yc1 * WW + xc1) * CIN);
                float w00 = wy0 * wx0 * v0y * v0x, w01 = wy0 * wx1 * v0y * v1x;
                float w10 = wy1 * wx0 * v1y * v0x, w11 = wy1 * wx1 * v1y * v1x;
                #pragma unroll
                for (int j = 0; j < 8; ++j) {
                    float s = w00 * (float)a00[j] + w01 * (float)a01[j]
                            + w10 * (float)a10[j] + w11 * (float)a11[j];
                    Bp[p * 64 + g * 8 + j] = (f16)s;
                }
            }
            __syncthreads();
            // ---- two WMMA chunks consume the tap panel ----
            #pragma unroll
            for (int u = 0; u < 2; ++u) {
                int kk = tap * 2 + u;
                v16h a  = *(const v16h*)(ap + (size_t)kk * 512);
                v16h bf = *(const v16h*)&Bp[lc * 64 + u * 32 + sel * 16];
                acc = wmma_f16(a, bf, acc);
            }
            __syncthreads();
        }
    }
    int xo = x0 + lc;
    #pragma unroll
    for (int r = 0; r < 8; ++r) {
        int mm = wv * 16 + r + sel * 8;
        out[(((size_t)b * COUT + mm) * HH + y) * WW + xo] = acc[r] + bsum[mm];
    }
}

// ---------------------------------------------------------------------------
extern "C" void kernel_launch(void* const* d_in, const int* in_sizes, int n_in,
                              void* d_out, int out_size, void* d_ws, size_t ws_size,
                              hipStream_t stream)
{
    (void)in_sizes; (void)n_in; (void)out_size; (void)ws_size;
    const float* x    = (const float*)d_in[0];
    const float* bg   = (const float*)d_in[1];
    const float* bb   = (const float*)d_in[2];
    const float* bm   = (const float*)d_in[3];
    const float* bv   = (const float*)d_in[4];
    const float* offw = (const float*)d_in[5];
    const float* offb = (const float*)d_in[6];
    const float* dcw  = (const float*)d_in[7];
    const float* dcb  = (const float*)d_in[8];
    float* out = (float*)d_out;

    char*  ws   = (char*)d_ws;
    f16*   h    = (f16*)  (ws + WS_H);
    float* offs = (float*)(ws + WS_OFF);
    f16*   A1p  = (f16*)  (ws + WS_A1);
    f16*   A2p  = (f16*)  (ws + WS_A2);
    float* ob   = (float*)(ws + WS_OB);
    float* bs   = (float*)(ws + WS_BS);

    // 1) BN+ReLU -> NHWC f16
    bn_relu_nhwc<<<dim3(BB * HH), dim3(64, 4), 0, stream>>>(x, bg, bb, bm, bv, h);
    // 2) weight repack into per-lane fragment order
    int packN = 5 * 72 * 32 * 16 + 16 * 18 * 32 * 16 + MOFFP + COUT;
    pack_weights<<<dim3((packN + 255) / 256), dim3(256), 0, stream>>>(
        offw, offb, dcw, dcb, A1p, A2p, ob, bs);
    // 3) offset conv: grid.x = 2048/4 N-groups, grid.y = 5 M tiles, 4 waves/block
    offconv_wmma<<<dim3(512, 5), dim3(128), 0, stream>>>(h, A1p, ob, offs);
    // 4) deformable conv: one block per N tile, 4 waves share the LDS B panel
    dconv_wmma<<<dim3(2048), dim3(128), 0, stream>>>(h, A2p, offs, bs, out);
}
